// HierarchicalDeceptionModel_6038724018277
// MI455X (gfx1250) — matrix operand order repaired and checked by reading, not compiled, not run on hardware
//
#include <hip/hip_runtime.h>
#include <hip/hip_bf16.h>
#include <stdint.h>

#define BB 64
#define TT 256
#define HH 128
#define NG 512

typedef __attribute__((ext_vector_type(16))) __bf16 v16bf;
typedef __attribute__((ext_vector_type(8)))  float  v8f;

static __device__ __forceinline__ float sigf(float x) { return 1.0f / (1.0f + __expf(-x)); }

#if __has_builtin(__builtin_amdgcn_tensor_load_to_lds)
#define TDM_AVAILABLE 1
#else
#define TDM_AVAILABLE 0
#endif

// ---------------------------------------------------------------------------
// TDM: async-DMA one contiguous slab (32768 x 4B = 128KB) global -> LDS.
// D# built per CDNA5 ISA: 1-D tile, data_size=4B,
// tile_dim0 = tensor_dim0 = 32768, no padding/iterate/multicast.
// ---------------------------------------------------------------------------
static __device__ __forceinline__ void tdm_load_slab(const float* gsrc, void* ldst)
{
#if TDM_AVAILABLE
    typedef __attribute__((ext_vector_type(4))) unsigned int u32x4;
    typedef __attribute__((ext_vector_type(4))) int          i32x4;
    typedef __attribute__((ext_vector_type(8))) int          i32x8;
    uint64_t ga = (uint64_t)(uintptr_t)gsrc;
    uint32_t la = (uint32_t)(uintptr_t)ldst;          // low 32 bits = LDS byte offset
    u32x4 g0;
    g0[0] = 1u;                                        // count=1, no gather
    g0[1] = la;                                        // lds_addr
    g0[2] = (uint32_t)ga;                              // global_addr[31:0]
    g0[3] = (uint32_t)((ga >> 32) & 0x1FFFFFFu) | (2u << 30);   // addr[56:32] | type=2
    i32x8 g1;
    g1[0] = (int)(2u << 16);                           // data_size = 4B
    g1[1] = (int)(32768u << 16);                       // tensor_dim0[15:0] = 32768
    g1[2] = (int)(1u << 16);                           // tensor_dim0 hi = 0; tensor_dim1 = 1
    g1[3] = (int)(32768u << 16);                       // tile_dim0 = 32768
    g1[4] = 0;                                         // tile_dim1/2 unused
    g1[5] = 32768;                                     // tensor_dim0_stride lo32
    g1[6] = 0;
    g1[7] = 0;
    i32x4 gz = {0, 0, 0, 0};
#if defined(__clang_major__) && (__clang_major__ >= 23)
    i32x8 gz8 = {0, 0, 0, 0, 0, 0, 0, 0};
    __builtin_amdgcn_tensor_load_to_lds(g0, g1, gz, gz, gz8, 0);
#else
    __builtin_amdgcn_tensor_load_to_lds(g0, g1, gz, gz, 0);
#endif
#endif
}

// ---------------------------------------------------------------------------
// One-time packing kernels: fp32 -> bf16 (activations), fp32 -> [512, K] bf16
// (weights) so the hot WMMA loops do pure 16B loads with no conversion VALU.
// ---------------------------------------------------------------------------
__global__ __launch_bounds__(256)
void pack_act_kernel(const float* __restrict__ in, __bf16* __restrict__ out, int n4)
{
    int i = blockIdx.x * 256 + threadIdx.x;
    if (i >= n4) return;
    float4 v = reinterpret_cast<const float4*>(in)[i];
    union { __bf16 h[4]; uint2 u; } o;
    o.h[0] = (__bf16)v.x; o.h[1] = (__bf16)v.y;
    o.h[2] = (__bf16)v.z; o.h[3] = (__bf16)v.w;
    reinterpret_cast<uint2*>(out)[i] = o.u;
}

// w: [512, K] fp32 -> wt: [512, K] bf16
__global__ __launch_bounds__(256)
void pack_wt_kernel(const float* __restrict__ w, __bf16* __restrict__ wt, int K)
{
    int i = blockIdx.x * 256 + threadIdx.x;     // over 512*K, n fastest, stride-K accesses
    int n = i & 511;
    int k = i >> 9;
    if (k >= K) return;
    wt[(size_t)n * K + k] = (__bf16)w[(size_t)n * K + k];
}

// ---------------------------------------------------------------------------
// Input-projection GEMM: xg[t*B+b, n] = A[b*T+t, :] . W^T + bih[n] + bhh[n]
// A: [B*T, Din] bf16.  WT: [512, Din] bf16 (row n, k contiguous).  xg: [T,B,512] f32.
// One wave = one 16x16 tile over full K; all operands are 16B b128 loads.
// ---------------------------------------------------------------------------
__global__ __launch_bounds__(128)
void xg_gemm_kernel(const __bf16* __restrict__ A, int Din,
                    const __bf16* __restrict__ WT,
                    const float* __restrict__ bih,
                    const float* __restrict__ bhh,
                    float* __restrict__ xg)
{
    const int lane = threadIdx.x & 31;
    const int w    = threadIdx.x >> 5;
    const int tm   = blockIdx.x;           // M tile (0..1023)
    const int tn   = blockIdx.y * 4 + w;   // N tile (0..31)
    const int hiH  = (lane >= 16) ? 1 : 0;
    const int l16  = lane & 15;
    const int n0   = tn * 16;

    float bias = bih[n0 + l16] + bhh[n0 + l16];
    v8f acc;
#pragma unroll
    for (int r = 0; r < 8; ++r) acc[r] = bias;

    const int arow = tm * 16 + l16;
    const __bf16* Arow = A + (size_t)arow * Din;
    const int nk = Din >> 5;

    for (int kk = 0; kk < nk; ++kk) {
        const __bf16* ap = Arow + kk * 32 + hiH * 8;
        __builtin_prefetch(ap + 64, 0, 1);   // global_prefetch_b8: next k-steps
        union { uint4 u; __bf16 h[8]; } a0, a1, b0, b1;
        a0.u = *reinterpret_cast<const uint4*>(ap);
        a1.u = *reinterpret_cast<const uint4*>(ap + 16);
        const __bf16* bp = WT + (size_t)(n0 + l16) * Din + kk * 32 + hiH * 8;
        b0.u = *reinterpret_cast<const uint4*>(bp);
        b1.u = *reinterpret_cast<const uint4*>(bp + 16);
        v16bf a, b;
#pragma unroll
        for (int e = 0; e < 8; ++e) {
            a[e] = a0.h[e]; a[8 + e] = a1.h[e];
            b[e] = b0.h[e]; b[8 + e] = b1.h[e];
        }
        acc = __builtin_amdgcn_wmma_f32_16x16x32_bf16(false, a, false, b,
                                                      (short)0, acc, false, false);
    }
#pragma unroll
    for (int r = 0; r < 8; ++r) {
        int m    = tm * 16 + r + hiH * 8;
        int bidx = m >> 8;       // /T
        int t    = m & 255;      // %T
        xg[((size_t)(t * BB + bidx)) * NG + n0 + l16] = acc[r];
    }
}

// ---------------------------------------------------------------------------
// LSTM recurrence: one workgroup (16 waves) runs one direction for all 256
// steps. WhhT fragments register-resident (128 VGPRs); h state ping-pongs
// through a 32KB LDS bf16 double buffer; c lives in VGPRs.
// Modes 0/1: next step's 128KB xg slab is TDM-prefetched into an LDS double
// buffer (s_wait_tensorcnt + barrier hides L2 latency behind the WMMA chain).
// mode: 0 = forward, 1 = full reverse (msg backward), 2 = ragged reverse (ctx,
// per-row gather -> direct global loads).
// out: [B, T, 256] bf16, written at column offset dirOff.
// Dynamic LDS: 32KB h ping-pong [+ 2 x 128KB xg slabs for modes 0/1].
// ---------------------------------------------------------------------------
__global__ __launch_bounds__(512)
void lstm_rec_kernel(const float* __restrict__ xg,
                     const __bf16* __restrict__ WhhT,   // [512, 128] bf16
                     __bf16* __restrict__ out, int dirOff, int mode,
                     const int* __restrict__ lens)
{
    extern __shared__ char smem[];
    typedef __bf16 HS[BB][HH];
    HS* h_s = (HS*)smem;                       // [2][BB][HH] ping-pong h state
    float* xg_s = (float*)(smem + 2 * BB * HH * 2);   // [2][32768] slabs (modes 0/1)

    const int tid  = threadIdx.x;
    const int lane = tid & 31;
    const int w    = tid >> 5;
    const int jt   = w & 7;
    const int mh   = w >> 3;
    const int hiH  = (lane >= 16) ? 1 : 0;
    const int l16  = lane & 15;
    const int j    = jt * 16 + l16;            // h column 0..127

    const bool useTdm = TDM_AVAILABLE && (mode != 2);

    for (int i = tid; i < 2 * BB * HH; i += 512)
        ((unsigned short*)smem)[i] = 0;

    // WhhT fragments: bfr[gate][kstep]; lane = column; K in table order.
    v16bf bfr[4][4];
#pragma unroll
    for (int g = 0; g < 4; ++g)
#pragma unroll
        for (int kk = 0; kk < 4; ++kk) {
            const __bf16* bp = WhhT + (size_t)(g * HH + jt * 16 + l16) * 128 + kk * 32 + hiH * 8;
            union { uint4 u; __bf16 h[8]; } lo, hi;
            lo.u = *reinterpret_cast<const uint4*>(bp);
            hi.u = *reinterpret_cast<const uint4*>(bp + 16);
#pragma unroll
            for (int e = 0; e < 8; ++e) { bfr[g][kk][e] = lo.h[e]; bfr[g][kk][8 + e] = hi.h[e]; }
        }

    int lenv[2][8];
#pragma unroll
    for (int mi = 0; mi < 2; ++mi)
#pragma unroll
        for (int r = 0; r < 8; ++r)
            lenv[mi][r] = lens[(mh * 2 + mi) * 16 + r + hiH * 8];

    v8f cst[2];
#pragma unroll
    for (int mi = 0; mi < 2; ++mi)
#pragma unroll
        for (int r = 0; r < 8; ++r) cst[mi][r] = 0.0f;

    // Preload slab for step 0 (modes 0/1): TDM from wave 0 only (EXEC-ignored op).
    if (useTdm && w == 0) {
        int t0 = (mode == 0) ? 0 : (TT - 1);
        tdm_load_slab(xg + (size_t)t0 * BB * NG, xg_s);
#if TDM_AVAILABLE
        __builtin_amdgcn_s_wait_tensorcnt(0);
#endif
    }
    __syncthreads();

    for (int s = 0; s < TT; ++s) {
        const int pr = s & 1;
        const int pw = pr ^ 1;
        // Kick off async TDM for the next step's slab into the other buffer.
        if (useTdm && w == 0 && (s + 1) < TT) {
            int tn1 = (mode == 0) ? (s + 1) : (TT - 2 - s);
            tdm_load_slab(xg + (size_t)tn1 * BB * NG, xg_s + (size_t)((s + 1) & 1) * 32768);
        }
        const float* slab = xg_s + (size_t)(s & 1) * 32768;

        for (int mi = 0; mi < 2; ++mi) {
            const int mt   = mh * 2 + mi;
            const int arow = mt * 16 + l16;
            v16bf afr[4];
#pragma unroll
            for (int kk = 0; kk < 4; ++kk) {
                const int kb = kk * 32 + hiH * 8;
                union { uint4 u; __bf16 h[8]; } lo, hi;
                lo.u = *reinterpret_cast<const uint4*>(&h_s[pr][arow][kb]);
                hi.u = *reinterpret_cast<const uint4*>(&h_s[pr][arow][kb + 16]);
#pragma unroll
                for (int e = 0; e < 8; ++e) { afr[kk][e] = lo.h[e]; afr[kk][8 + e] = hi.h[e]; }
            }
            v8f acc[4];
#pragma unroll
            for (int g = 0; g < 4; ++g) {
#pragma unroll
                for (int r = 0; r < 8; ++r) {
                    int brow = mt * 16 + r + hiH * 8;
                    if (useTdm) {
                        acc[g][r] = slab[brow * NG + g * HH + j];
                    } else {
                        int tin;
                        if (mode == 0)      tin = s;
                        else if (mode == 1) tin = TT - 1 - s;
                        else { int L = lenv[mi][r]; tin = L - 1 - s; if (tin < 0) tin = 0; }
                        acc[g][r] = xg[((size_t)(tin * BB + brow)) * NG + g * HH + j];
                    }
                }
#pragma unroll
                for (int kk = 0; kk < 4; ++kk)
                    acc[g] = __builtin_amdgcn_wmma_f32_16x16x32_bf16(
                        false, afr[kk], false, bfr[g][kk], (short)0, acc[g], false, false);
            }
#pragma unroll
            for (int r = 0; r < 8; ++r) {
                float iv = sigf(acc[0][r]);
                float fv = sigf(acc[1][r]);
                float gv = tanhf(acc[2][r]);
                float ov = sigf(acc[3][r]);
                float c  = fv * cst[mi][r] + iv * gv;
                cst[mi][r] = c;
                float h = ov * tanhf(c);
                __bf16 hb = (__bf16)h;
                int brow = mt * 16 + r + hiH * 8;
                h_s[pw][brow][j] = hb;
                int tout; bool doSt = true;
                if (mode == 0)      tout = s;
                else if (mode == 1) tout = TT - 1 - s;
                else { int L = lenv[mi][r]; if (s >= L) { doSt = false; tout = 0; } else tout = L - 1 - s; }
                if (doSt) out[((size_t)brow * TT + tout) * 256 + dirOff + j] = hb;
            }
        }
        // Ensure the async slab for step s+1 landed, then release all waves.
        if (useTdm && w == 0) {
#if TDM_AVAILABLE
            __builtin_amdgcn_s_wait_tensorcnt(0);
#endif
        }
        __syncthreads();
    }
}

// ---------------------------------------------------------------------------
// Head: z = ReLU(LN(concat(ctx, meta) @ fiW^T + fib)); h = ReLU(z@W1^T+b1);
// logits = h@W2^T + b2; gather valid (b,t) rows row-major into d_out.
// ---------------------------------------------------------------------------
__global__ __launch_bounds__(256)
void head_kernel(const __bf16* __restrict__ ctx, const float* __restrict__ meta,
                 const int* __restrict__ lens,
                 const float* __restrict__ fiW, const float* __restrict__ fib,
                 const float* __restrict__ fig, const float* __restrict__ fibeta,
                 const float* __restrict__ W1,  const float* __restrict__ b1,
                 const float* __restrict__ W2,  const float* __restrict__ b2,
                 float* __restrict__ out)
{
    __shared__ float zin[276];
    __shared__ float red[256];
    __shared__ float z2[256];
    __shared__ float hbuf[128];
    __shared__ float stats[2];
    const int t = blockIdx.x, b = blockIdx.y, tid = threadIdx.x;

    zin[tid] = (float)ctx[((size_t)b * TT + t) * 256 + tid];
    if (tid < 20) zin[256 + tid] = meta[((size_t)b * TT + t) * 20 + tid];
    __syncthreads();

    float y = fib[tid];
    {
        const float* wr = fiW + (size_t)tid * 276;
        for (int k = 0; k < 276; ++k) y += wr[k] * zin[k];
    }
    red[tid] = y; __syncthreads();
    for (int off = 128; off > 0; off >>= 1) { if (tid < off) red[tid] += red[tid + off]; __syncthreads(); }
    if (tid == 0) stats[0] = red[0] * (1.0f / 256.0f);
    __syncthreads();
    float d = y - stats[0];
    red[tid] = d * d; __syncthreads();
    for (int off = 128; off > 0; off >>= 1) { if (tid < off) red[tid] += red[tid + off]; __syncthreads(); }
    if (tid == 0) stats[1] = red[0] * (1.0f / 256.0f);
    __syncthreads();
    float zz = d * rsqrtf(stats[1] + 1e-5f) * fig[tid] + fibeta[tid];
    z2[tid] = zz > 0.0f ? zz : 0.0f;
    __syncthreads();
    if (tid < 128) {
        float h = b1[tid];
        const float* wr = W1 + (size_t)tid * 256;
        for (int k = 0; k < 256; ++k) h += wr[k] * z2[k];
        hbuf[tid] = h > 0.0f ? h : 0.0f;
    }
    __syncthreads();
    if (tid < 2) {
        int L = lens[b];
        if (t < L) {
            float o = b2[tid];
            const float* wr = W2 + (size_t)tid * 128;
            for (int k = 0; k < 128; ++k) o += wr[k] * hbuf[k];
            int base = 0;
            for (int bb = 0; bb < b; ++bb) base += lens[bb];
            out[(size_t)(base + t) * 2 + tid] = o;
        }
    }
}

// ---------------------------------------------------------------------------
extern "C" void kernel_launch(void* const* d_in, const int* in_sizes, int n_in,
                              void* d_out, int out_size, void* d_ws, size_t ws_size,
                              hipStream_t stream)
{
    (void)in_sizes; (void)n_in; (void)out_size; (void)ws_size;
    const float* emb  = (const float*)d_in[0];
    const float* meta = (const float*)d_in[1];
    const int*   lens = (const int*)d_in[2];
    // 6 LSTM directions × {Wih, Whh, bih, bhh}: 0:l0f 1:l0b 2:l1f 3:l1b 4:ctxf 5:ctxb
    const float* P[24];
    for (int i = 0; i < 24; ++i) P[i] = (const float*)d_in[3 + i];
    const float* fiW    = (const float*)d_in[27];
    const float* fib    = (const float*)d_in[28];
    const float* fig    = (const float*)d_in[29];
    const float* fibeta = (const float*)d_in[30];
    const float* W1 = (const float*)d_in[31];
    const float* b1 = (const float*)d_in[32];
    const float* W2 = (const float*)d_in[33];
    const float* b2 = (const float*)d_in[34];

    char* ws = (char*)d_ws;
    size_t off = 0;
    auto alloc = [&](size_t bytes) -> void* {
        void* p = ws + off;
        off += (bytes + 255) & ~(size_t)255;
        return p;
    };

    float*  xgA  = (float*)alloc((size_t)TT * BB * NG * 4);       // 32 MB [T,B,512] f32
    float*  xgB  = (float*)alloc((size_t)TT * BB * NG * 4);       // 32 MB
    __bf16* embb = (__bf16*)alloc((size_t)BB * TT * 768 * 2);     // 24 MB
    __bf16* h0b  = (__bf16*)alloc((size_t)BB * TT * 256 * 2);     //  8 MB [B,T,256] bf16
    __bf16* h1b  = (__bf16*)alloc((size_t)BB * TT * 256 * 2);
    __bf16* cxb  = (__bf16*)alloc((size_t)BB * TT * 256 * 2);
    __bf16* wihT[6];
    __bf16* whhT[6];
    const int DinOf[6] = {768, 768, 256, 256, 256, 256};
    for (int i = 0; i < 6; ++i) wihT[i] = (__bf16*)alloc((size_t)DinOf[i] * 512 * 2);
    for (int i = 0; i < 6; ++i) whhT[i] = (__bf16*)alloc((size_t)128 * 512 * 2);

    // ---- pack activations + weights to bf16 (weights keep [512, K] layout) ----
    {
        int n4 = BB * TT * 768 / 4;
        pack_act_kernel<<<dim3((n4 + 255) / 256), dim3(256), 0, stream>>>(emb, embb, n4);
    }
    for (int i = 0; i < 6; ++i) {
        int K = DinOf[i];
        pack_wt_kernel<<<dim3(512 * K / 256), dim3(256), 0, stream>>>(P[i * 4 + 0], wihT[i], K);
        pack_wt_kernel<<<dim3(512 * 128 / 256), dim3(256), 0, stream>>>(P[i * 4 + 1], whhT[i], 128);
    }

    dim3 gg(1024, 8), gb(128);
    dim3 rg(1), rb(512);
    const size_t ldsTdm  = (size_t)2 * BB * HH * 2 + (size_t)2 * 32768 * 4;  // 288 KB
    const size_t ldsBase = (size_t)2 * BB * HH * 2;                          //  32 KB

    // Stage 1: message LSTM layer 0 (Din = 768)
    xg_gemm_kernel<<<gg, gb, 0, stream>>>(embb, 768, wihT[0], P[2],  P[3],  xgA);
    xg_gemm_kernel<<<gg, gb, 0, stream>>>(embb, 768, wihT[1], P[6],  P[7],  xgB);
    lstm_rec_kernel<<<rg, rb, ldsTdm, stream>>>(xgA, whhT[0], h0b, 0,   0, lens);
    lstm_rec_kernel<<<rg, rb, ldsTdm, stream>>>(xgB, whhT[1], h0b, 128, 1, lens);

    // Stage 2: message LSTM layer 1 (Din = 256)
    xg_gemm_kernel<<<gg, gb, 0, stream>>>(h0b, 256, wihT[2], P[10], P[11], xgA);
    xg_gemm_kernel<<<gg, gb, 0, stream>>>(h0b, 256, wihT[3], P[14], P[15], xgB);
    lstm_rec_kernel<<<rg, rb, ldsTdm, stream>>>(xgA, whhT[2], h1b, 0,   0, lens);
    lstm_rec_kernel<<<rg, rb, ldsTdm, stream>>>(xgB, whhT[3], h1b, 128, 1, lens);

    // Stage 3: context LSTM (Din = 256); backward is ragged-reversed
    xg_gemm_kernel<<<gg, gb, 0, stream>>>(h1b, 256, wihT[4], P[18], P[19], xgA);
    xg_gemm_kernel<<<gg, gb, 0, stream>>>(h1b, 256, wihT[5], P[22], P[23], xgB);
    lstm_rec_kernel<<<rg, rb, ldsTdm, stream>>>(xgA, whhT[4], cxb, 0,   0, lens);
    lstm_rec_kernel<<<rg, rb, ldsBase, stream>>>(xgB, whhT[5], cxb, 128, 2, lens);

    // Head + valid-row gather
    head_kernel<<<dim3(256, 64), 256, 0, stream>>>(cxb, meta, lens, fiW, fib, fig, fibeta,
                                                   W1, b1, W2, b2, (float*)d_out);
}
